// MultiscaleGlobalWaveGNNMoE_59064390255202
// MI455X (gfx1250) — matrix operand order repaired and checked by reading, not compiled, not run on hardware
//
#include <hip/hip_runtime.h>

#define H 128

typedef __attribute__((ext_vector_type(16))) __bf16 v16bf;
typedef __attribute__((ext_vector_type(8)))  __bf16 v8bf;
typedef __attribute__((ext_vector_type(8)))  float  v8f;
typedef __attribute__((ext_vector_type(4)))  float  f32x4;

__device__ __forceinline__ __bf16 f2bf(float f) {
  return static_cast<__bf16>(f);   // native v_cvt on gfx1250
}
__device__ __forceinline__ unsigned short bf16bits(float f) {
  __bf16 h = static_cast<__bf16>(f);
  return __builtin_bit_cast(unsigned short, h);
}

__device__ __forceinline__ v8f wmma_bf16(v16bf a, v16bf b, v8f c) {
  return __builtin_amdgcn_wmma_f32_16x16x32_bf16(
      /*neg_a=*/false, a, /*neg_b=*/false, b,
      /*c_mod=*/(short)0, c, /*reuse_a=*/false, /*reuse_b=*/false);
}

// 16x32 bf16 A-fragment from a bf16 (ushort) feature row, K contiguous.
// ISA layout: lane<16 holds K {base..base+7, base+16..base+23} of row (lane&15);
// caller passes base = 32*j + (lane>=16 ? 8 : 0).
__device__ __forceinline__ v16bf load_afrag_bf16(const unsigned short* __restrict__ row, int base) {
  v8bf lo = *(const v8bf*)(row + base);
  v8bf hi = *(const v8bf*)(row + base + 16);
  return __builtin_shufflevector(lo, hi, 0, 1, 2, 3, 4, 5, 6, 7,
                                 8, 9, 10, 11, 12, 13, 14, 15);
}

// Same from an f32 row (used for the f32 agg buffers), converting on the fly.
__device__ __forceinline__ v16bf load_afrag_f32(const float* __restrict__ row, int base) {
  f32x4 a0 = *(const f32x4*)(row + base);
  f32x4 a1 = *(const f32x4*)(row + base + 4);
  f32x4 a2 = *(const f32x4*)(row + base + 16);
  f32x4 a3 = *(const f32x4*)(row + base + 20);
  v16bf r;
#pragma unroll
  for (int i = 0; i < 4; ++i) {
    r[i]      = f2bf(a0[i]);
    r[4 + i]  = f2bf(a1[i]);
    r[8 + i]  = f2bf(a2[i]);
    r[12 + i] = f2bf(a3[i]);
  }
  return r;
}

__device__ __forceinline__ v16bf load_bfrag(const char* fbase, int lane, int frag) {
  return *(const v16bf*)(fbase + frag * 1024 + lane * 32);
}

// Stage W (Ktot x 128, row-major f32) into LDS as bf16 B-fragments.
// Fragment f = j*8 + t covers K in [32j,32j+32), N in [16t,16t+16).
// ISA dense-B: lane<16 holds K 0..7,16..23 of column (lane&15), lane>=16 holds K 8..15,24..31.
__device__ __forceinline__ void stage_frags(const float* __restrict__ W, char* dst,
                                            int nfrag, int tid, int nthreads) {
  for (int idx = tid; idx < nfrag * 512; idx += nthreads) {
    int f = idx >> 9, w = idx & 511;
    int lane = w >> 4, e = w & 15;
    int j = f >> 3, t = f & 7;
    int K = 32 * j + ((lane >> 4) << 3) + (e & 7) + ((e >> 3) << 4);
    int n = 16 * t + (lane & 15);
    *(unsigned short*)(dst + f * 1024 + lane * 32 + e * 2) = bf16bits(W[K * H + n]);
  }
}

// Double-buffered 8-tile WMMA sweep for one k-step: keeps two B-fragment
// register sets alive so the next tile's ds_loads overlap the current WMMA
// (backend can emit partial s_wait_dscnt instead of dscnt==0 every time).
__device__ __forceinline__ void wmma_sweep(v8f acc[8], v16bf a, const char* fbase,
                                           int lane, int j) {
  v16bf bcur = load_bfrag(fbase, lane, j * 8);
#pragma unroll
  for (int t = 0; t < 8; ++t) {
    v16bf bnext = (t < 7) ? load_bfrag(fbase, lane, j * 8 + t + 1) : bcur;
    acc[t] = wmma_bf16(a, bcur, acc[t]);
    bcur = bnext;
  }
}

__global__ void zero_kernel(float* __restrict__ p, long n) {
  size_t i = (size_t)blockIdx.x * blockDim.x + threadIdx.x;
  size_t stride = (size_t)gridDim.x * blockDim.x;
  for (; i < (size_t)n; i += stride) p[i] = 0.0f;
}

// One-time f32 -> bf16 conversion of node features (no conversion VALU in the
// per-edge hot loop; halves the L2 gather bytes).
__global__ void cvt_bf16_kernel(const float* __restrict__ src,
                                unsigned short* __restrict__ dst, long n) {
  size_t i = (size_t)blockIdx.x * blockDim.x + threadIdx.x;
  size_t stride = (size_t)gridDim.x * blockDim.x;
  for (; i < (size_t)n; i += stride) dst[i] = bf16bits(src[i]);
}

// ---------------------------------------------------------------------------
// Edge message kernel: 8 waves/block, each wave a 16-edge x 128-col tile.
// msgs = relu([x[src]|x[dst]|ea] @ W1 + b1) @ W2 + b2 ; scatter-add to agg[dst].
// ---------------------------------------------------------------------------
__global__ __launch_bounds__(256, 1)
void msg_kernel(const unsigned short* __restrict__ xbf, const int* __restrict__ ei,
                const float* __restrict__ ea, const float* __restrict__ W1,
                const float* __restrict__ b1, const float* __restrict__ W2,
                const float* __restrict__ b2, float* __restrict__ agg, int E) {
  extern __shared__ float4 smem4[];
  char* smem = (char*)smem4;
  char*  w1f = smem;                       // 64 frags * 1KB = 65536
  char*  w2f = smem + 65536;               // 32 frags * 1KB = 32768
  float* wea = (float*)(smem + 98304);     // W1 rows 256..258 (3*128 f32)
  float* b1s = (float*)(smem + 99840);     // 128 f32
  float* b2s = (float*)(smem + 100352);    // 128 f32
  char*  stg_all = smem + 100864;          // 8 waves * 16*128 bf16 = 32768

  int tid = threadIdx.x;
  stage_frags(W1, w1f, 64, tid, 256);      // K=256 -> 8 k-steps * 8 tiles
  stage_frags(W2, w2f, 32, tid, 256);      // K=128 -> 4 k-steps * 8 tiles
  if (tid < 128) {
    int c = tid;
    wea[c]       = W1[256 * H + c];
    wea[128 + c] = W1[257 * H + c];
    wea[256 + c] = W1[258 * H + c];
    b1s[c] = b1[c];
    b2s[c] = b2[c];
  }
  __syncthreads();

  int wave = tid >> 5, lane = tid & 31;
  int hi = lane >> 4, col = lane & 15;
  int tile = (blockIdx.x * 8 + wave) * 16;

  // A row (edge) carried by this lane
  int aclamp = min(tile + col, E - 1);
  const unsigned short* srcp = xbf + (size_t)ei[aclamp] * H;
  const unsigned short* dstp = xbf + (size_t)ei[E + aclamp] * H;

  // Accumulator seed: b1 + rank-3 edge-attr contribution (VALU)
  float e0[8], e1[8], e2[8];
#pragma unroll
  for (int r = 0; r < 8; ++r) {
    int em = min(tile + r + hi * 8, E - 1);
    e0[r] = ea[em * 3 + 0];
    e1[r] = ea[em * 3 + 1];
    e2[r] = ea[em * 3 + 2];
  }
  v8f acc[8];
#pragma unroll
  for (int t = 0; t < 8; ++t) {
    int n = t * 16 + col;
    float bn = b1s[n], w0 = wea[n], w1v = wea[128 + n], w2v = wea[256 + n];
#pragma unroll
    for (int r = 0; r < 8; ++r)
      acc[t][r] = fmaf(e0[r], w0, fmaf(e1[r], w1v, fmaf(e2[r], w2v, bn)));
  }

  // Layer 1: K = 256 (src features then dst features), all-bf16 loads
#pragma unroll
  for (int j = 0; j < 8; ++j) {
    const unsigned short* rowp = (j < 4) ? srcp : dstp;
    int base = (j & 3) * 32 + hi * 8;
    v16bf a = load_afrag_bf16(rowp, base);
    wmma_sweep(acc, a, w1f, lane, j);
  }

  // ReLU -> bf16 staging (per-wave LDS region; cross-lane transpose for layer 2)
  unsigned short* stg = (unsigned short*)(stg_all + wave * 4096);
#pragma unroll
  for (int t = 0; t < 8; ++t)
#pragma unroll
    for (int r = 0; r < 8; ++r) {
      int mm = r + hi * 8, n = t * 16 + col;
      stg[mm * H + n] = bf16bits(fmaxf(acc[t][r], 0.0f));
    }

  // Layer 2: K = 128
  v8f acc2[8];
#pragma unroll
  for (int t = 0; t < 8; ++t) {
    float bn = b2s[t * 16 + col];
#pragma unroll
    for (int r = 0; r < 8; ++r) acc2[t][r] = bn;
  }
#pragma unroll
  for (int j = 0; j < 4; ++j) {
    v16bf a = load_afrag_bf16(stg + col * H, j * 32 + hi * 8);
    wmma_sweep(acc2, a, w2f, lane, j);
  }

  // Scatter-add into agg[target]
#pragma unroll
  for (int r = 0; r < 8; ++r) {
    int em = tile + r + hi * 8;
    if (em < E) {
      float* dptr = agg + (size_t)ei[E + em] * H;
#pragma unroll
      for (int t = 0; t < 8; ++t)
        __hip_atomic_fetch_add(dptr + t * 16 + col, acc2[t][r],
                               __ATOMIC_RELAXED, __HIP_MEMORY_SCOPE_AGENT);
    }
  }
}

// ---------------------------------------------------------------------------
// Node kernel 1: h1u = relu([x|aggL|aggM|aggG] @ Wu1 + bu1) -> bf16 ws buffer
// ---------------------------------------------------------------------------
__global__ __launch_bounds__(256, 1)
void node1_kernel(const unsigned short* __restrict__ xbf,
                  const float* __restrict__ aggA, const float* __restrict__ aggB,
                  const float* __restrict__ aggC,
                  const float* __restrict__ Wu1, const float* __restrict__ bu1,
                  unsigned short* __restrict__ h1, int N) {
  extern __shared__ float4 smem4[];
  char* smem = (char*)smem4;
  char*  wf  = smem;                       // 128 frags * 1KB (K=512)
  float* bs  = (float*)(smem + 131072);

  int tid = threadIdx.x;
  stage_frags(Wu1, wf, 128, tid, 256);
  if (tid < 128) bs[tid] = bu1[tid];
  __syncthreads();

  int wave = tid >> 5, lane = tid & 31;
  int hi = lane >> 4, col = lane & 15;
  int tile = (blockIdx.x * 8 + wave) * 16;
  int node = min(tile + col, N - 1);
  const unsigned short* px = xbf + (size_t)node * H;
  const float* p1 = aggA + (size_t)node * H;
  const float* p2 = aggB + (size_t)node * H;
  const float* p3 = aggC + (size_t)node * H;

  v8f acc[8];
#pragma unroll
  for (int t = 0; t < 8; ++t) {
    float bn = bs[t * 16 + col];
#pragma unroll
    for (int r = 0; r < 8; ++r) acc[t][r] = bn;
  }
#pragma unroll
  for (int j = 0; j < 16; ++j) {
    int base = (j & 3) * 32 + hi * 8;
    v16bf a;
    if (j < 4) {
      a = load_afrag_bf16(px, base);
    } else {
      const float* rowp = (j < 8) ? p1 : (j < 12) ? p2 : p3;
      a = load_afrag_f32(rowp, base);
    }
    wmma_sweep(acc, a, wf, lane, j);
  }
#pragma unroll
  for (int t = 0; t < 8; ++t)
#pragma unroll
    for (int r = 0; r < 8; ++r) {
      int nd = tile + r + hi * 8;
      if (nd < N)
        h1[(size_t)nd * H + t * 16 + col] = bf16bits(fmaxf(acc[t][r], 0.0f));
    }
}

// ---------------------------------------------------------------------------
// Node kernel 2: gate = sigmoid(u@Wg+bg); upd = h1u@Wu2+bu2;
// out = LayerNorm(gate*upd + (1-gate)*x) * gamma + beta
// ---------------------------------------------------------------------------
__global__ __launch_bounds__(256, 1)
void node2_kernel(const float* __restrict__ x, const unsigned short* __restrict__ xbf,
                  const float* __restrict__ aggA, const float* __restrict__ aggB,
                  const float* __restrict__ aggC,
                  const unsigned short* __restrict__ h1,
                  const float* __restrict__ Wg, const float* __restrict__ bg,
                  const float* __restrict__ Wu2, const float* __restrict__ bu2,
                  const float* __restrict__ gamma, const float* __restrict__ beta,
                  float* __restrict__ out, int N) {
  extern __shared__ float4 smem4[];
  char* smem = (char*)smem4;
  char*  wgf  = smem;                          // 128 frags (K=512) = 131072
  char*  wu2f = smem + 131072;                 // 32 frags (K=128) = 32768
  float* bgs  = (float*)(smem + 163840);
  float* b2s  = (float*)(smem + 164352);
  float* gms  = (float*)(smem + 164864);
  float* bts  = (float*)(smem + 165376);
  float* gstg_all = (float*)(smem + 165888);   // 8 waves * 16*128 f32 = 65536
  float* ostg_all = (float*)(smem + 231424);   // 8 waves * 16*128 f32 = 65536

  int tid = threadIdx.x;
  stage_frags(Wg, wgf, 128, tid, 256);
  stage_frags(Wu2, wu2f, 32, tid, 256);
  if (tid < 128) {
    bgs[tid] = bg[tid];
    b2s[tid] = bu2[tid];
    gms[tid] = gamma[tid];
    bts[tid] = beta[tid];
  }
  __syncthreads();

  int wave = tid >> 5, lane = tid & 31;
  int hi = lane >> 4, col = lane & 15;
  int tile = (blockIdx.x * 8 + wave) * 16;
  int node = min(tile + col, N - 1);
  const unsigned short* px = xbf + (size_t)node * H;
  const float* p1 = aggA + (size_t)node * H;
  const float* p2 = aggB + (size_t)node * H;
  const float* p3 = aggC + (size_t)node * H;

  // ---- gate pre-activation: u @ Wg + bg, K = 512 ----
  v8f accg[8];
#pragma unroll
  for (int t = 0; t < 8; ++t) {
    float bn = bgs[t * 16 + col];
#pragma unroll
    for (int r = 0; r < 8; ++r) accg[t][r] = bn;
  }
#pragma unroll
  for (int j = 0; j < 16; ++j) {
    int base = (j & 3) * 32 + hi * 8;
    v16bf a;
    if (j < 4) {
      a = load_afrag_bf16(px, base);
    } else {
      const float* rowp = (j < 8) ? p1 : (j < 12) ? p2 : p3;
      a = load_afrag_f32(rowp, base);
    }
    wmma_sweep(accg, a, wgf, lane, j);
  }
  float* gstg = gstg_all + wave * 2048;
#pragma unroll
  for (int t = 0; t < 8; ++t)
#pragma unroll
    for (int r = 0; r < 8; ++r)
      gstg[(r + hi * 8) * H + t * 16 + col] = accg[t][r];

  // ---- update layer 2: relu(h1u) @ Wu2 + bu2, K = 128 (A from bf16 ws) ----
  v8f acco[8];
#pragma unroll
  for (int t = 0; t < 8; ++t) {
    float bn = b2s[t * 16 + col];
#pragma unroll
    for (int r = 0; r < 8; ++r) acco[t][r] = bn;
  }
  const unsigned short* h1row = h1 + (size_t)node * H;
#pragma unroll
  for (int j = 0; j < 4; ++j) {
    v16bf a = load_afrag_bf16(h1row, j * 32 + hi * 8);
    wmma_sweep(acco, a, wu2f, lane, j);
  }

  // ---- gated residual combine into LDS (residual path uses f32 x) ----
  float* ostg = ostg_all + wave * 2048;
#pragma unroll
  for (int t = 0; t < 8; ++t)
#pragma unroll
    for (int r = 0; r < 8; ++r) {
      int mm = r + hi * 8, n = t * 16 + col;
      int nd = min(tile + mm, N - 1);
      float g = 1.0f / (1.0f + __expf(-gstg[mm * H + n]));
      float xv = x[(size_t)nd * H + n];
      ostg[mm * H + n] = g * acco[t][r] + (1.0f - g) * xv;
    }

  // ---- LayerNorm: lanes 0..15, one row each (WMMA done; divergence OK) ----
  if (lane < 16) {
    int nd = tile + lane;
    if (nd < N) {
      const float* row = ostg + lane * H;
      float s = 0.0f, ss = 0.0f;
#pragma unroll 8
      for (int c = 0; c < H; ++c) { float v = row[c]; s += v; ss += v * v; }
      float mu  = s * (1.0f / H);
      float var = ss * (1.0f / H) - mu * mu;
      float inv = rsqrtf(var + 1e-5f);
      float* op = out + (size_t)nd * H;
#pragma unroll 8
      for (int c = 0; c < H; ++c)
        op[c] = (row[c] - mu) * inv * gms[c] + bts[c];
    }
  }
}

extern "C" void kernel_launch(void* const* d_in, const int* in_sizes, int n_in,
                              void* d_out, int out_size, void* d_ws, size_t ws_size,
                              hipStream_t stream) {
  const float* x   = (const float*)d_in[0];
  const int*   eiL = (const int*)d_in[1];
  const float* eaL = (const float*)d_in[2];
  const int*   eiM = (const int*)d_in[3];
  const float* eaM = (const float*)d_in[4];
  const int*   eiG = (const int*)d_in[5];
  const float* eaG = (const float*)d_in[6];
  const float* Wa1 = (const float*)d_in[7],  *ba1 = (const float*)d_in[8];
  const float* Wa2 = (const float*)d_in[9],  *ba2 = (const float*)d_in[10];
  const float* Wb1 = (const float*)d_in[11], *bb1 = (const float*)d_in[12];
  const float* Wb2 = (const float*)d_in[13], *bb2 = (const float*)d_in[14];
  const float* Wc1 = (const float*)d_in[15], *bc1 = (const float*)d_in[16];
  const float* Wc2 = (const float*)d_in[17], *bc2 = (const float*)d_in[18];
  const float* Wg  = (const float*)d_in[19], *bg  = (const float*)d_in[20];
  const float* Wu1 = (const float*)d_in[21], *bu1 = (const float*)d_in[22];
  const float* Wu2 = (const float*)d_in[23], *bu2 = (const float*)d_in[24];
  const float* gamma = (const float*)d_in[25], *beta = (const float*)d_in[26];

  int N  = in_sizes[0] / H;
  int EL = in_sizes[1] / 2;
  int EM = in_sizes[3] / 2;
  int EG = in_sizes[5] / 2;

  char* ws = (char*)d_ws;
  float* aggL = (float*)ws;
  float* aggM = aggL + (size_t)N * H;
  float* aggG = aggM + (size_t)N * H;
  unsigned short* h1  = (unsigned short*)(aggG + (size_t)N * H);
  unsigned short* xbf = h1 + (size_t)N * H;

  // Zero the scatter-add accumulators (graph replays don't reset ws), then
  // pre-convert node features to bf16 once.
  zero_kernel<<<2048, 256, 0, stream>>>((float*)ws, 3L * N * H);
  cvt_bf16_kernel<<<2048, 256, 0, stream>>>(x, xbf, (long)N * H);

  const size_t MSG_LDS = 133632;   // W1f 64K + W2f 32K + misc + 32K staging
  const size_t N1_LDS  = 131584;   // Wu1 frags 128K + bias
  const size_t N2_LDS  = 296960;   // Wg 128K + Wu2 32K + params + gate/out staging

  msg_kernel<<<(EL + 127) / 128, 256, MSG_LDS, stream>>>(xbf, eiL, eaL, Wa1, ba1, Wa2, ba2, aggL, EL);
  msg_kernel<<<(EM + 127) / 128, 256, MSG_LDS, stream>>>(xbf, eiM, eaM, Wb1, bb1, Wb2, bb2, aggM, EM);
  msg_kernel<<<(EG + 127) / 128, 256, MSG_LDS, stream>>>(xbf, eiG, eaG, Wc1, bc1, Wc2, bc2, aggG, EG);

  int nb = (N + 127) / 128;
  node1_kernel<<<nb, 256, N1_LDS, stream>>>(xbf, aggL, aggM, aggG, Wu1, bu1, h1, N);
  node2_kernel<<<nb, 256, N2_LDS, stream>>>(x, xbf, aggL, aggM, aggG, h1, Wg, bg, Wu2, bu2,
                                            gamma, beta, (float*)d_out, N);
}
